// AlphaNet_47596827574819
// MI455X (gfx1250) — matrix operand order repaired and checked by reading, not compiled, not run on hardware
//
#include <hip/hip_runtime.h>
#include <hip/hip_bf16.h>
#include <math.h>

typedef __attribute__((ext_vector_type(16))) _Float16 v16h;
typedef __attribute__((ext_vector_type(8)))  _Float16 v8h;
typedef __attribute__((ext_vector_type(8)))  float    v8f;

#define NCH   9
#define TT    30
#define DD    10
#define NW    3
#define NIN   702
#define KPAD  712      // padded K stride (f16 elems); 702 used + zeros
#define ELEMS 128      // batch rows per workgroup (= blockDim.x)
#define NKT   22       // 22 * 32 = 704 >= 702

__global__ __launch_bounds__(ELEMS, 1)
void alphanet_fused(const float* __restrict__ X,
                    const float* __restrict__ BN1H, const float* __restrict__ BN1N,
                    const float* __restrict__ BN2H, const float* __restrict__ BN2N,
                    const float* __restrict__ W1g,  const float* __restrict__ B1g,
                    const float* __restrict__ W2g,  const float* __restrict__ B2g,
                    float* __restrict__ Out, int Btot)
{
    __shared__ _Float16 lds_f[ELEMS * KPAD];   // feature tile, f16
    __shared__ _Float16 lds_w1[32 * KPAD];     // W1 padded to 32 x 704, f16
    __shared__ float s1[117], t1[117];         // BN1 scale/shift
    __shared__ float s2[351], t2[351];         // BN2 scale/shift

    const int tid = threadIdx.x;

    // ---- stage BN1 affine (y = x*s + t) ----
    for (int i = tid; i < 117; i += ELEMS) {
        float g, bb, rm, rv;
        if (i < 72) { int L = i / 36, c = i % 36; const float* P = BN1H + L * 144;
            g = P[c]; bb = P[36 + c]; rm = P[72 + c]; rv = P[108 + c]; }
        else { int j = i - 72; int L = j / 9, c = j % 9; const float* P = BN1N + L * 36;
            g = P[c]; bb = P[9 + c]; rm = P[18 + c]; rv = P[27 + c]; }
        float s = g / sqrtf(rv + 1e-5f);
        s1[i] = s; t1[i] = bb - rm * s;
    }
    // ---- stage BN2 affine ----
    for (int i = tid; i < 351; i += ELEMS) {
        float g, bb, rm, rv;
        if (i < 216) { int L = i / 36, c = i % 36; const float* P = BN2H + L * 144;
            g = P[c]; bb = P[36 + c]; rm = P[72 + c]; rv = P[108 + c]; }
        else { int j = i - 216; int L = j / 9, c = j % 9; const float* P = BN2N + L * 36;
            g = P[c]; bb = P[9 + c]; rm = P[18 + c]; rv = P[27 + c]; }
        float s = g / sqrtf(rv + 1e-5f);
        s2[i] = s; t2[i] = bb - rm * s;
    }
    // ---- stage W1 as f16 (zero-padded to 32 x KPAD) ----
    for (int i = tid; i < 32 * KPAD; i += ELEMS) {
        int n = i / KPAD, k = i - n * KPAD;
        float v = (n < 30 && k < NIN) ? W1g[n * NIN + k] : 0.0f;
        lds_w1[i] = (_Float16)v;
    }
    __syncthreads();

    // ================= feature extraction (1 thread = 1 batch row) =========
    const int e = tid;
    const long b = (long)blockIdx.x * ELEMS + e;
    _Float16* frow = lds_f + e * KPAD;
    for (int k = NIN; k < KPAD; ++k) frow[k] = (_Float16)0.0f;  // K pad zeros

    if (b < (long)Btot) {
        const float* xb = X + b * (NCH * TT);
        for (int w = 0; w < NW; ++w) {
            float xv[NCH][DD];
            float vs[NCH];
            #pragma unroll
            for (int n = 0; n < NCH; ++n) {
                const float* p = xb + n * TT + w * DD;
                float s = 0.0f, dec = 0.0f;
                #pragma unroll
                for (int d = 0; d < DD; ++d) {
                    float v = p[d]; xv[n][d] = v;
                    s += v; dec += v * ((float)(d + 1) * (1.0f / 55.0f));
                }
                float mu = s * 0.1f;
                float x0 = xv[n][0];
                float ret = (xv[n][DD - 1] - x0) / x0 - 1.0f;
                float v2 = 0.0f;
                #pragma unroll
                for (int d = 0; d < DD; ++d) { float c = xv[n][d] - mu; xv[n][d] = c; v2 += c * c; }
                vs[n] = v2;
                float stdu = sqrtf(v2 * (1.0f / 9.0f));
                float zsc = mu / stdu;
                int kw;
                kw = 216 + n * 3 + w; frow[kw] = (_Float16)(stdu * s1[72 + n]  + t1[72 + n]);
                kw = 243 + n * 3 + w; frow[kw] = (_Float16)(zsc  * s1[81 + n]  + t1[81 + n]);
                kw = 270 + n * 3 + w; frow[kw] = (_Float16)(ret  * s1[90 + n]  + t1[90 + n]);
                kw = 297 + n * 3 + w; frow[kw] = (_Float16)(dec  * s1[99 + n]  + t1[99 + n]);
                kw = 324 + n * 3 + w; frow[kw] = (_Float16)(mu   * s1[108 + n] + t1[108 + n]);
            }
            int p = 0;
            #pragma unroll
            for (int j = 0; j < NCH; ++j) {
                #pragma unroll
                for (int k2 = j + 1; k2 < NCH; ++k2) {
                    float dot = 0.0f;
                    #pragma unroll
                    for (int d = 0; d < DD; ++d) dot += xv[j][d] * xv[k2][d];
                    float corr = dot / sqrtf(vs[j] * vs[k2]);   // (dot/D)/(std_j*std_k)
                    float cov  = dot * 0.1f;
                    frow[p * 3 + w]       = (_Float16)(corr * s1[p]      + t1[p]);
                    frow[108 + p * 3 + w] = (_Float16)(cov  * s1[36 + p] + t1[36 + p]);
                    ++p;
                }
            }
        }
        // pooling over 3 windows (avg/max/min of BN1'd values) + BN2
        for (int ch = 0; ch < 117; ++ch) {
            int k0, pb, c, Cs;
            if (ch < 36)      { k0 = ch * 3;              pb = 0;   c = ch;      Cs = 36; }
            else if (ch < 72) { k0 = 108 + (ch - 36) * 3; pb = 108; c = ch - 36; Cs = 36; }
            else { int s = (ch - 72) / 9; c = (ch - 72) % 9;
                   k0 = 216 + (ch - 72) * 3; pb = 216 + s * 27; Cs = 9; }
            float v0 = (float)frow[k0], v1 = (float)frow[k0 + 1], v2 = (float)frow[k0 + 2];
            float av = (v0 + v1 + v2) * (1.0f / 3.0f);
            float mx = fmaxf(v0, fmaxf(v1, v2));
            float mn = fminf(v0, fminf(v1, v2));
            int ca = pb + c;
            frow[351 + ca]          = (_Float16)(av * s2[ca]          + t2[ca]);
            frow[351 + ca + Cs]     = (_Float16)(mx * s2[ca + Cs]     + t2[ca + Cs]);
            frow[351 + ca + 2 * Cs] = (_Float16)(mn * s2[ca + 2 * Cs] + t2[ca + 2 * Cs]);
        }
    } else {
        for (int k = 0; k < NIN; ++k) frow[k] = (_Float16)0.0f;
    }
    __syncthreads();

    // ================= WMMA GEMM: f(128x704) @ W1^T(704x32) ================
    const int wave = tid >> 5;
    const int lane = tid & 31;
    const int m16  = lane & 15;
    const int kh   = lane >> 4;
    const float b2v  = B2g[0];
    const int   n1   = 16 + m16;
    const float b1v0 = B1g[m16];
    const float b1v1 = (n1 < 30) ? B1g[n1] : 0.0f;
    const float w2v0 = W2g[m16];
    const float w2v1 = (n1 < 30) ? W2g[n1] : 0.0f;

    for (int mt = 0; mt < 2; ++mt) {
        const int mbase = wave * 32 + mt * 16;
        const _Float16* arow  = lds_f  + (mbase + m16) * KPAD;
        const _Float16* brow0 = lds_w1 + m16 * KPAD;
        const _Float16* brow1 = lds_w1 + (16 + m16) * KPAD;
        v8f c0 = {}, c1 = {};
        for (int kt = 0; kt < NKT; ++kt) {
            // A frag (16x32 f16): lane m16, two 8-elem K runs at kh*8 and 16+kh*8
            const int ka = kt * 32 + kh * 8;
            v16h a, bm0, bm1;
            ((v8h*)&a)[0] = *(const v8h*)(arow + ka);
            ((v8h*)&a)[1] = *(const v8h*)(arow + ka + 16);
            // B frag (32x16 f16): lane col=m16, two 8-elem K runs at kh*8 and 16+kh*8
            const int kb = kt * 32 + kh * 8;
            ((v8h*)&bm0)[0] = *(const v8h*)(brow0 + kb);
            ((v8h*)&bm0)[1] = *(const v8h*)(brow0 + kb + 16);
            ((v8h*)&bm1)[0] = *(const v8h*)(brow1 + kb);
            ((v8h*)&bm1)[1] = *(const v8h*)(brow1 + kb + 16);
            c0 = __builtin_amdgcn_wmma_f32_16x16x32_f16(false, a, false, bm0,
                                                        (short)0, c0, false, false);
            c1 = __builtin_amdgcn_wmma_f32_16x16x32_f16(false, a, false, bm1,
                                                        (short)0, c1, false, false);
        }
        // epilogue: bias + ReLU + W2 dot, reduce across the 16 neuron-lanes
        float part[8];
        #pragma unroll
        for (int r = 0; r < 8; ++r) {
            float h0 = fmaxf(c0[r] + b1v0, 0.0f) * w2v0;
            float h1 = fmaxf(c1[r] + b1v1, 0.0f) * w2v1;  // cols >=30 are 0
            part[r] = h0 + h1;
        }
        #pragma unroll
        for (int msk = 1; msk < 16; msk <<= 1) {
            #pragma unroll
            for (int r = 0; r < 8; ++r)
                part[r] += __shfl_xor(part[r], msk, 16);
        }
        if (m16 == 0) {
            long ebase = (long)blockIdx.x * ELEMS + mbase + kh * 8;
            #pragma unroll
            for (int r = 0; r < 8; ++r)
                if (ebase + r < (long)Btot) Out[ebase + r] = part[r] + b2v;
        }
    }
}

extern "C" void kernel_launch(void* const* d_in, const int* in_sizes, int n_in,
                              void* d_out, int out_size, void* d_ws, size_t ws_size,
                              hipStream_t stream) {
    const float* X    = (const float*)d_in[0];
    const float* BN1H = (const float*)d_in[1];
    const float* BN1N = (const float*)d_in[2];
    const float* BN2H = (const float*)d_in[3];
    const float* BN2N = (const float*)d_in[4];
    const float* W1g  = (const float*)d_in[5];
    const float* B1g  = (const float*)d_in[6];
    const float* W2g  = (const float*)d_in[7];
    const float* B2g  = (const float*)d_in[8];
    float* Out = (float*)d_out;

    const int Btot = in_sizes[0] / (NCH * TT);
    const int grid = (Btot + ELEMS - 1) / ELEMS;
    alphanet_fused<<<grid, ELEMS, 0, stream>>>(X, BN1H, BN1N, BN2H, BN2N,
                                               W1g, B1g, W2g, B2g, Out, Btot);
}
